// LightGCL_6545530159207
// MI455X (gfx1250) — matrix operand order repaired and checked by reading, compiled, not run on hardware
//
#include <hip/hip_runtime.h>

// ---------------- problem constants (from reference) ----------------
#define N_U   100000
#define N_I   100000
#define DIM   64
#define NQ    5
#define N_E   1000000
#define BB    2048
#define INV_TEMP 5.0f          // 1/0.2
#define SCALE_LOG2E 7.2134752044448170f   // (1/TEMP) * log2(e), folded into f16 A

typedef __attribute__((ext_vector_type(16))) _Float16 v16h;
typedef __attribute__((ext_vector_type(8)))  _Float16 v8h;
typedef __attribute__((ext_vector_type(8)))  float    v8f;

// ---------------- workspace layout (float offsets) ----------------
// Zeroed region first (contiguous), then buffers that are fully written.
static constexpr size_t OFF_ZU1   = 0;                               // layer-1 Z_u  (N_U*64)
static constexpr size_t OFF_ZI1   = OFF_ZU1 + (size_t)N_U * DIM;     // layer-1 Z_i  (N_I*64)
static constexpr size_t OFF_VTE1  = OFF_ZI1 + (size_t)N_I * DIM;     // vt@E_i_0   (5*64)
static constexpr size_t OFF_UTE1  = OFF_VTE1 + NQ * DIM;             // ut@E_u_0
static constexpr size_t OFF_VTE2  = OFF_UTE1 + NQ * DIM;             // vt@Zi1
static constexpr size_t OFF_UTE2  = OFF_VTE2 + NQ * DIM;             // ut@Zu1
static constexpr size_t OFF_SEXPU = OFF_UTE2 + NQ * DIM;             // rowwise sum(exp(s_u)) (B)
static constexpr size_t OFF_SEXPI = OFF_SEXPU + BB;                  // rowwise sum(exp(s_i)) (B)
static constexpr size_t OFF_SCAL  = OFF_SEXPI + BB;                  // [pos,neg,bpr,reg] sums (8)
static constexpr size_t ZERO_CNT  = OFF_SCAL + 8;
static constexpr size_t OFF_EUN   = ZERO_CNT;                        // E_u_sum (fully written)
static constexpr size_t OFF_EIN   = OFF_EUN + (size_t)N_U * DIM;     // E_i_sum
static constexpr size_t OFF_GUS   = OFF_EIN + (size_t)N_I * DIM;     // G_u_n[row_ids] (B*64)
static constexpr size_t OFF_GIS   = OFF_GUS + (size_t)BB * DIM;      // G_i_n[col_ids]
static constexpr size_t F_END     = OFF_GIS + (size_t)BB * DIM;
static constexpr size_t HALF_BASE = ((F_END * 4 + 127) / 128) * 128; // byte offset, 128B aligned
static constexpr size_t HOFF_EUNH = 0;                               // half offsets
static constexpr size_t HOFF_EINH = HOFF_EUNH + (size_t)N_U * DIM;
static constexpr size_t HOFF_GUSH = HOFF_EINH + (size_t)N_I * DIM;
static constexpr size_t HOFF_GISH = HOFF_GUSH + (size_t)BB * DIM;

// ---------------- kernels ----------------

__global__ void k_zero(float* __restrict__ w, size_t n) {
  size_t i  = (size_t)blockIdx.x * blockDim.x + threadIdx.x;
  size_t st = (size_t)gridDim.x * blockDim.x;
  for (; i < n; i += st) w[i] = 0.0f;
}

// out[q][d] = sum_r T[q*n+r] * E[r*64+d]    (T is Q x n row-major)
__global__ void k_reduce_qt(const float* __restrict__ T, const float* __restrict__ E,
                            float* __restrict__ out, int n) {
  int d    = threadIdx.x & 63;
  int grp  = threadIdx.x >> 6;                // 4 row-groups per block
  int step = gridDim.x * 4;
  float acc[NQ] = {0.f, 0.f, 0.f, 0.f, 0.f};
  for (int r = blockIdx.x * 4 + grp; r < n; r += step) {
    float e = E[(size_t)r * DIM + d];
#pragma unroll
    for (int q = 0; q < NQ; ++q) acc[q] = fmaf(T[(size_t)q * n + r], e, acc[q]);
  }
#pragma unroll
  for (int q = 0; q < NQ; ++q) atomicAdd(&out[q * DIM + d], acc[q]);
}

// Edge-parallel SpMM, both directions at once. 16 lanes/edge, 4 dims/lane.
// Zu[rows[e]] += v*Ei[cols[e]] ; Zi[cols[e]] += v*Eu[rows[e]]
__global__ void k_spmm(const int* __restrict__ rows, const int* __restrict__ cols,
                       const float* __restrict__ vals,
                       const float* __restrict__ Eu, const float* __restrict__ Ei,
                       float* __restrict__ Zu, float* __restrict__ Zi, int ne) {
  size_t t = (size_t)blockIdx.x * blockDim.x + threadIdx.x;
  size_t e = t >> 4;
  if (e >= (size_t)ne) return;
  int l = (int)(t & 15);
  int r = rows[e], c = cols[e];
  float v = vals[e];
  float4 xi = ((const float4*)(Ei + (size_t)c * DIM))[l];
  float* zu = Zu + (size_t)r * DIM + l * 4;
  atomicAdd(zu + 0, v * xi.x); atomicAdd(zu + 1, v * xi.y);
  atomicAdd(zu + 2, v * xi.z); atomicAdd(zu + 3, v * xi.w);
  float4 xu = ((const float4*)(Eu + (size_t)r * DIM))[l];
  float* zi = Zi + (size_t)c * DIM + l * 4;
  atomicAdd(zi + 0, v * xu.x); atomicAdd(zi + 1, v * xu.y);
  atomicAdd(zi + 2, v * xu.z); atomicAdd(zi + 3, v * xu.w);
}

__global__ void k_add(const float* __restrict__ a, const float* __restrict__ b,
                      float* __restrict__ o, size_t n) {
  size_t i = (size_t)blockIdx.x * blockDim.x + threadIdx.x;
  if (i < n) o[i] = a[i] + b[i];
}

__global__ void k_tohalf(const float* __restrict__ s, _Float16* __restrict__ d, size_t n) {
  size_t i = (size_t)blockIdx.x * blockDim.x + threadIdx.x;
  if (i < n) d[i] = (_Float16)s[i];
}

// G_sel[b] = E0[ids[b]] + muls[ids[b]] @ (tE1 + tE2); f16 mirror pre-scaled by
// (1/TEMP)*log2(e) so the GEMM kernel can use exp2 directly on WMMA output.
__global__ void k_gather_g(const int* __restrict__ ids, const float* __restrict__ E0,
                           const float* __restrict__ muls, const float* __restrict__ tE1,
                           const float* __restrict__ tE2, float* __restrict__ Gsel,
                           _Float16* __restrict__ GselH) {
  int t = blockIdx.x * blockDim.x + threadIdx.x;
  if (t >= BB * DIM) return;
  int b = t >> 6, d = t & 63;
  int r = ids[b];
  float g = E0[(size_t)r * DIM + d];
#pragma unroll
  for (int q = 0; q < NQ; ++q)
    g = fmaf(muls[(size_t)r * NQ + q], tE1[q * DIM + d] + tE2[q * DIM + d], g);
  Gsel[t]  = g;
  GselH[t] = (_Float16)(g * SCALE_LOG2E);
}

__global__ void k_sumsq(const float* __restrict__ x, size_t n, float* __restrict__ out) {
  size_t i  = (size_t)blockIdx.x * blockDim.x + threadIdx.x;
  size_t st = (size_t)gridDim.x * blockDim.x;
  float a = 0.f;
  for (; i < n; i += st) { float v = x[i]; a = fmaf(v, v, a); }
#pragma unroll
  for (int m = 1; m < 32; m <<= 1) a += __shfl_xor(a, m, 32);
  if ((threadIdx.x & 31) == 0) atomicAdd(out, a);
}

// Fused WMMA GEMM + exp + row-sum:  sumExp[m] += sum_n exp( (G[m]·E[n]) / TEMP )
// One wave owns a 16-row M stripe; A fragment resident, B streamed.
// Two adjacent 16x16 N tiles per iteration (4 WMMAs), double-buffered loads.
// A is pre-scaled by (1/TEMP)*log2(e), so sum(exp) = sum(exp2(C)); logits are
// far from the exp2 denorm threshold, so use raw v_exp_f32.
__global__ void __launch_bounds__(256) k_gemm_exp(const _Float16* __restrict__ Gh,
                                                  const _Float16* __restrict__ Eh,
                                                  float* __restrict__ sumExp, int nTiles) {
  const int lane = threadIdx.x & 31;
  const int wv   = threadIdx.x >> 5;
  const int m0   = blockIdx.x << 4;
  const int half = lane >> 4;      // upper/lower 16-lane group
  const int l16  = lane & 15;

  // A fragment: lane L<16 -> K{0..15}; lane L>=16 -> K{16..31} (+32 for a1)
  const v8h* Ar = (const v8h*)(Gh + (size_t)(m0 + l16) * DIM);
  v16h a0 = __builtin_shufflevector(Ar[0 + 2 * half], Ar[1 + 2 * half],
                                    0,1,2,3,4,5,6,7,8,9,10,11,12,13,14,15);
  v16h a1 = __builtin_shufflevector(Ar[4 + 2 * half], Ar[5 + 2 * half],
                                    0,1,2,3,4,5,6,7,8,9,10,11,12,13,14,15);

  // B fragment loader: lane = column N; halves = K 0..15 (lanes 0-15) / 16..31 (16-31)
  auto ldB = [&](int t, v16h& x0, v16h& x1) {
    const v16h* Br = (const v16h*)(Eh + ((size_t)t * 16 + l16) * DIM);
    x0 = Br[half];
    x1 = Br[2 + half];
  };

  float acc[8] = {0.f,0.f,0.f,0.f,0.f,0.f,0.f,0.f};
  const int nPairs = nTiles >> 1;               // 2 adjacent tiles per iteration
  const int wstep  = gridDim.y * 8;
  int p = blockIdx.y * 8 + wv;

  v16h b0a, b1a, b0b, b1b;                      // current pair (double buffer front)
  if (p < nPairs) { ldB(2 * p, b0a, b1a); ldB(2 * p + 1, b0b, b1b); }
  while (p < nPairs) {
    const int pn = p + wstep;
    v16h n0a = b0a, n1a = b1a, n0b = b0b, n1b = b1b;
    if (pn < nPairs) { ldB(2 * pn, n0a, n1a); ldB(2 * pn + 1, n0b, n1b); }
    // branch-free clamped prefetch two strides ahead (global_prefetch_b8)
    int pf = p + 2 * wstep; if (pf >= nPairs) pf = p;
    __builtin_prefetch(Eh + ((size_t)(2 * pf) * 16 + l16) * DIM, 0, 0);

    v8f ca = {0.f,0.f,0.f,0.f,0.f,0.f,0.f,0.f};
    ca = __builtin_amdgcn_wmma_f32_16x16x32_f16(false, a0, false, b0a, (short)0, ca, false, false);
    ca = __builtin_amdgcn_wmma_f32_16x16x32_f16(false, a1, false, b1a, (short)0, ca, false, false);
    v8f cb = {0.f,0.f,0.f,0.f,0.f,0.f,0.f,0.f};
    cb = __builtin_amdgcn_wmma_f32_16x16x32_f16(false, a0, false, b0b, (short)0, cb, false, false);
    cb = __builtin_amdgcn_wmma_f32_16x16x32_f16(false, a1, false, b1b, (short)0, cb, false, false);
#pragma unroll
    for (int v = 0; v < 8; ++v)
      acc[v] += __builtin_amdgcn_exp2f(ca[v]) + __builtin_amdgcn_exp2f(cb[v]);

    b0a = n0a; b1a = n1a; b0b = n0b; b1b = n1b;
    p = pn;
  }
  // C layout: VGPR v, lanes 0-15 -> row m0+v; lanes 16-31 -> row m0+8+v; column = lane%16
#pragma unroll
  for (int v = 0; v < 8; ++v) {
    float s = acc[v];
    s += __shfl_xor(s, 1, 16);
    s += __shfl_xor(s, 2, 16);
    s += __shfl_xor(s, 4, 16);
    s += __shfl_xor(s, 8, 16);
    if (l16 == 0) atomicAdd(&sumExp[m0 + v + (half << 3)], s);
  }
}

// pos_score, BPR, InfoNCE-neg log terms. 16 lanes per batch element.
__global__ void k_tail(const int* __restrict__ row_ids, const int* __restrict__ col_ids,
                       const int* __restrict__ pos, const int* __restrict__ neg,
                       const float* __restrict__ GUS, const float* __restrict__ GIS,
                       const float* __restrict__ EUN, const float* __restrict__ EIN,
                       const float* __restrict__ seU, const float* __restrict__ seI,
                       float* __restrict__ scal) {
  int t = blockIdx.x * blockDim.x + threadIdx.x;
  int b = t >> 4, l = t & 15;
  if (b >= BB) return;
  int ru = row_ids[b], ci = col_ids[b], pp = pos[b], nn = neg[b];
  float4 gu = ((const float4*)(GUS + (size_t)b * DIM))[l];
  float4 eu = ((const float4*)(EUN + (size_t)ru * DIM))[l];
  float4 gi = ((const float4*)(GIS + (size_t)b * DIM))[l];
  float4 ei = ((const float4*)(EIN + (size_t)ci * DIM))[l];
  float4 ep = ((const float4*)(EIN + (size_t)pp * DIM))[l];
  float4 en = ((const float4*)(EIN + (size_t)nn * DIM))[l];
  float pu = gu.x*eu.x + gu.y*eu.y + gu.z*eu.z + gu.w*eu.w;
  float pi = gi.x*ei.x + gi.y*ei.y + gi.z*ei.z + gi.w*ei.w;
  float dd = eu.x*(ep.x-en.x) + eu.y*(ep.y-en.y) + eu.z*(ep.z-en.z) + eu.w*(ep.w-en.w);
#pragma unroll
  for (int m = 1; m < 16; m <<= 1) {
    pu += __shfl_xor(pu, m, 16);
    pi += __shfl_xor(pi, m, 16);
    dd += __shfl_xor(dd, m, 16);
  }
  if (l == 0) {
    float posc = fminf(fmaxf(pu * INV_TEMP, -5.f), 5.f)
               + fminf(fmaxf(pi * INV_TEMP, -5.f), 5.f);
    atomicAdd(&scal[0], posc);
    atomicAdd(&scal[1], logf(seU[b] + 1e-8f) + logf(seI[b] + 1e-8f));
    float sp = fmaxf(-dd, 0.f) + log1pf(__expf(-fabsf(dd)));  // -log_sigmoid(dd)
    atomicAdd(&scal[2], sp);
  }
}

__global__ void k_final(const float* __restrict__ scal, float* __restrict__ out) {
  float posv = scal[0] / (float)BB;
  float negv = scal[1] / (float)BB;
  float bpr  = scal[2] / (float)BB;
  float reg  = scal[3];
  float cl   = negv - posv;
  out[0] = bpr + 0.2f * cl + 1e-7f * reg;
  out[1] = bpr;
  out[2] = 0.2f * cl;
}

// ---------------- launcher ----------------
extern "C" void kernel_launch(void* const* d_in, const int* in_sizes, int n_in,
                              void* d_out, int out_size, void* d_ws, size_t ws_size,
                              hipStream_t stream) {
  (void)in_sizes; (void)n_in; (void)out_size; (void)ws_size;
  const float* E_u_0   = (const float*)d_in[0];
  const float* E_i_0   = (const float*)d_in[1];
  const float* vals    = (const float*)d_in[2];
  const float* u_mul_s = (const float*)d_in[3];
  const float* v_mul_s = (const float*)d_in[4];
  const float* ut      = (const float*)d_in[5];
  const float* vt      = (const float*)d_in[6];
  const int*   rows    = (const int*)d_in[7];
  const int*   cols    = (const int*)d_in[8];
  const int*   row_ids = (const int*)d_in[9];
  const int*   col_ids = (const int*)d_in[10];
  const int*   pos     = (const int*)d_in[11];
  const int*   neg     = (const int*)d_in[12];
  float* out = (float*)d_out;

  float* W = (float*)d_ws;
  float* ZU1 = W + OFF_ZU1;   float* ZI1 = W + OFF_ZI1;
  float* VTE1 = W + OFF_VTE1; float* UTE1 = W + OFF_UTE1;
  float* VTE2 = W + OFF_VTE2; float* UTE2 = W + OFF_UTE2;
  float* SEXPU = W + OFF_SEXPU; float* SEXPI = W + OFF_SEXPI;
  float* SCAL = W + OFF_SCAL;
  float* EUN = W + OFF_EUN;   float* EIN = W + OFF_EIN;
  float* GUS = W + OFF_GUS;   float* GIS = W + OFF_GIS;
  _Float16* HB   = (_Float16*)((char*)d_ws + HALF_BASE);
  _Float16* EUNH = HB + HOFF_EUNH;  _Float16* EINH = HB + HOFF_EINH;
  _Float16* GUSH = HB + HOFF_GUSH;  _Float16* GISH = HB + HOFF_GISH;

  const size_t ND = (size_t)N_U * DIM;  // == N_I*DIM

  // 0) zero accumulators (Z buffers, vtE/utE, exp-sums, scalar sums)
  k_zero<<<8192, 256, 0, stream>>>(W, ZERO_CNT);

  // 1) layer-1 low-rank reductions: vt@E_i_0, ut@E_u_0
  k_reduce_qt<<<192, 256, 0, stream>>>(vt, E_i_0, VTE1, N_I);
  k_reduce_qt<<<192, 256, 0, stream>>>(ut, E_u_0, UTE1, N_U);

  // 2) layer-1 SpMM -> ZU1, ZI1
  k_spmm<<<(N_E * 16) / 256, 256, 0, stream>>>(rows, cols, vals, E_u_0, E_i_0, ZU1, ZI1, N_E);

  // 3) E_sum after layer 1: EUN = E_u_0 + ZU1 ; EIN = E_i_0 + ZI1
  k_add<<<(int)(ND / 256), 256, 0, stream>>>(E_u_0, ZU1, EUN, ND);
  k_add<<<(int)(ND / 256), 256, 0, stream>>>(E_i_0, ZI1, EIN, ND);

  // 4) layer-2 low-rank reductions on layer-1 outputs
  k_reduce_qt<<<192, 256, 0, stream>>>(vt, ZI1, VTE2, N_I);
  k_reduce_qt<<<192, 256, 0, stream>>>(ut, ZU1, UTE2, N_U);

  // 5) layer-2 SpMM accumulates straight into E_sum (Z2 only feeds the sum)
  k_spmm<<<(N_E * 16) / 256, 256, 0, stream>>>(rows, cols, vals, ZU1, ZI1, EUN, EIN, N_E);

  // 6) f16 mirrors of E_sum for WMMA
  k_tohalf<<<(int)(ND / 256), 256, 0, stream>>>(EUN, EUNH, ND);
  k_tohalf<<<(int)(ND / 256), 256, 0, stream>>>(EIN, EINH, ND);

  // 7) batched G rows: G_u_n[row_ids], G_i_n[col_ids] (f32 + pre-scaled f16)
  k_gather_g<<<(BB * DIM) / 256, 256, 0, stream>>>(row_ids, E_u_0, u_mul_s, VTE1, VTE2, GUS, GUSH);
  k_gather_g<<<(BB * DIM) / 256, 256, 0, stream>>>(col_ids, E_i_0, v_mul_s, UTE1, UTE2, GIS, GISH);

  // 8) L2 regularization over parameters
  k_sumsq<<<2048, 256, 0, stream>>>(E_u_0, ND, &SCAL[3]);
  k_sumsq<<<2048, 256, 0, stream>>>(E_i_0, ND, &SCAL[3]);

  // 9) fused WMMA GEMM + exp row-sum (the 2 x 2048x100000x64 GEMMs, never stored)
  dim3 ggrid(BB / 16, 16);
  k_gemm_exp<<<ggrid, 256, 0, stream>>>(GUSH, EUNH, SEXPU, N_U / 16);
  k_gemm_exp<<<ggrid, 256, 0, stream>>>(GISH, EINH, SEXPI, N_I / 16);

  // 10) pos/neg/BPR accumulation, then final combine
  k_tail<<<(BB * 16) / 256, 256, 0, stream>>>(row_ids, col_ids, pos, neg,
                                              GUS, GIS, EUN, EIN, SEXPU, SEXPI, SCAL);
  k_final<<<1, 1, 0, stream>>>(SCAL, out);
}
